// CGCN_84026740179030
// MI455X (gfx1250) — matrix operand order repaired and checked by reading, not compiled, not run on hardware
//
#include <hip/hip_runtime.h>
#include <hip/hip_bf16.h>

// ---------------- problem constants (match reference) ----------------
#define NUM_NODES 2848
#define BGRAPH    64
#define NN        (BGRAPH * NUM_NODES)   // 182272 nodes
#define NE        (NN * 8)               // 1458176 edges
#define UDIM      64
#define EPSV      1e-5f

typedef __attribute__((ext_vector_type(16))) __bf16 v16bf;
typedef __attribute__((ext_vector_type(8)))  float  v8f;

// ---------------- generic helpers ----------------
__global__ void zero_f32_kernel(float* __restrict__ p, int n) {
    int i = blockIdx.x * blockDim.x + threadIdx.x;
    if (i < n) p[i] = 0.0f;
}

// deg[src[e]] += w[e]
__global__ void deg_kernel(const int* __restrict__ src, const float* __restrict__ w,
                           float* __restrict__ deg, int e) {
    int i = blockIdx.x * blockDim.x + threadIdx.x;
    if (i < e) atomicAdd(&deg[src[i]], w[i]);
}

// deg -> d^-1/2 in place
__global__ void dinv_kernel(float* __restrict__ deg, int n) {
    int i = blockIdx.x * blockDim.x + threadIdx.x;
    if (i < n) {
        float d = deg[i];
        deg[i] = (d > 0.0f) ? rsqrtf(fmaxf(d, EPSV)) : 0.0f;
    }
}

// norm_w[e] = -w[e] * dinv[src] * dinv[dst]
__global__ void normw_kernel(const int* __restrict__ src, const int* __restrict__ dst,
                             const float* __restrict__ w, const float* __restrict__ dinv,
                             float* __restrict__ nw, int e) {
    int i = blockIdx.x * blockDim.x + threadIdx.x;
    if (i < e) nw[i] = -w[i] * dinv[src[i]] * dinv[dst[i]];
}

// out[dst[e]*C + c] += nw[e] * in[src[e]*C + c]
// C consecutive threads share one edge -> coalesced gather & atomic scatter (L2-resident).
template <int C>
__global__ void prop_kernel(const float* __restrict__ in, float* __restrict__ out,
                            const int* __restrict__ src, const int* __restrict__ dst,
                            const float* __restrict__ nw) {
    int gid = blockIdx.x * blockDim.x + threadIdx.x;
    int e = gid / C;
    int c = gid % C;
    if (e < NE) {
        atomicAdd(&out[dst[e] * C + c], nw[e] * in[src[e] * C + c]);
    }
}

// ---------------- layer 1 (C_IN = 2, too skinny for WMMA) ----------------
__global__ void lin1_kernel(const float* __restrict__ x, const float* __restrict__ t1s,
                            const float* __restrict__ p2s, const float* __restrict__ W1,
                            const float* __restrict__ b1, float* __restrict__ out) {
    int gid = blockIdx.x * blockDim.x + threadIdx.x;
    if (gid >= NN * UDIM) return;
    int n = gid >> 6, c = gid & 63;
    float x0 = x[n * 2], x1 = x[n * 2 + 1];
    float t0 = t1s[n * 2], t1v = t1s[n * 2 + 1];
    float q0 = 2.0f * p2s[n * 2] - x0;
    float q1 = 2.0f * p2s[n * 2 + 1] - x1;
    float o = b1[c];
    o += x0 * W1[c]        + x1 * W1[64 + c];         // W1[0]
    o += t0 * W1[128 + c]  + t1v * W1[192 + c];       // W1[1]
    o += q0 * W1[256 + c]  + q1 * W1[320 + c];        // W1[2]
    out[gid] = o;
}

// ---------------- BatchNorm stats / apply ----------------
// stats: [0,64)=sum [64,128)=sumsq [128,192)=scale [192,256)=shift
__global__ void colstats_kernel(const float* __restrict__ in, float* __restrict__ stats) {
    __shared__ float ls[256], ls2[256];
    int c = threadIdx.x & 63, rg = threadIdx.x >> 6;
    float s = 0.0f, s2 = 0.0f;
    for (int r = blockIdx.x * 4 + rg; r < NN; r += gridDim.x * 4) {
        float v = in[r * 64 + c];
        s += v; s2 += v * v;
    }
    ls[threadIdx.x] = s; ls2[threadIdx.x] = s2;
    __syncthreads();
    if (rg == 0) {
        s  = ls[c]  + ls[64 + c]  + ls[128 + c]  + ls[192 + c];
        s2 = ls2[c] + ls2[64 + c] + ls2[128 + c] + ls2[192 + c];
        atomicAdd(&stats[c], s);
        atomicAdd(&stats[64 + c], s2);
    }
}

__global__ void bn_finalize_kernel(float* __restrict__ stats, const float* __restrict__ g,
                                   const float* __restrict__ be) {
    int c = threadIdx.x;
    if (c < 64) {
        float m  = stats[c] * (1.0f / (float)NN);
        float v  = stats[64 + c] * (1.0f / (float)NN) - m * m;
        float sc = g[c] * rsqrtf(v + EPSV);
        stats[128 + c] = sc;
        stats[192 + c] = be[c] - m * sc;
    }
}

__global__ void bn_apply_kernel(const float* __restrict__ in, float* __restrict__ out,
                                const float* __restrict__ stats) {
    int gid = blockIdx.x * blockDim.x + threadIdx.x;
    if (gid >= NN * UDIM) return;
    int c = gid & 63;
    float v = in[gid] * stats[128 + c] + stats[192 + c];
    out[gid] = v > 0.0f ? v : 0.0f;
}

// ---------------- weight fragment prep (f32 -> bf16, WMMA B lane layout) ----------------
// wf[((mat*2 + kt)*4 + nt)*512 + lane*16 + j] = bf16(W[k][K][n]),
//   K = kt*32 + 8*(lane/16) + 16*(j/8) + (j%8) , n = nt*16 + (lane%16)   (ISA 7.12.2 B layout)
__global__ void wfrag_kernel(const float* __restrict__ W2, const float* __restrict__ W3,
                             const float* __restrict__ W4, __bf16* __restrict__ wf) {
    int o = blockIdx.x * blockDim.x + threadIdx.x;
    if (o >= 9 * 4096) return;
    int j    = o & 15;
    int lane = (o >> 4) & 31;
    int nt   = (o >> 9) & 3;
    int kt   = (o >> 11) & 1;
    int mat  = o >> 12;                       // 0..8 = {W2,W3,W4} x {k=0,1,2}
    const float* W = (mat < 3) ? W2 : ((mat < 6) ? W3 : W4);
    int k = mat % 3;
    int K = kt * 32 + 8 * (lane >> 4) + 16 * (j >> 3) + (j & 7);
    int n = nt * 16 + (lane & 15);
    wf[o] = (__bf16)W[k * 4096 + K * 64 + n];
}

// ---------------- WMMA ChebConv GEMM: out = Tx0@W0 + Tx1@W1 + (2*p2 - Tx0)@W2 + b ----------
// One wave per 16-row tile; 2 K-tiles x 3 weights x 4 column tiles = 24 v_wmma_f32_16x16x32_bf16.
// Output written in place over p2 (all p2 reads feed the WMMAs before any store).
__global__ __launch_bounds__(256)
void cheb_gemm_kernel(const float* __restrict__ tx0, const float* __restrict__ tx1,
                      float* __restrict__ p2, const __bf16* __restrict__ wf,
                      const float* __restrict__ bias) {
    int lane = threadIdx.x & 31;
    int wave = threadIdx.x >> 5;
    int rt   = blockIdx.x * 8 + wave;     // exact grid: 11392 row tiles / 8 waves
    int lh   = lane >> 4;                 // half-wave select
    int nlo  = lane & 15;
    int row  = rt * 16 + nlo;             // A-matrix row owned by this lane

    v8f acc[4];
#pragma unroll
    for (int nt = 0; nt < 4; ++nt) {
        float bv = bias[nt * 16 + nlo];
        v8f a;
#pragma unroll
        for (int r = 0; r < 8; ++r) a[r] = bv;
        acc[nt] = a;
    }

#pragma unroll
    for (int k = 0; k < 3; ++k) {
#pragma unroll
        for (int kt = 0; kt < 2; ++kt) {
            // A fragment: element j -> K = kt*32 + 8*lh + 16*(j/8) + (j%8)  (ISA 7.12.2)
            int off = row * 64 + kt * 32 + lh * 8;
            v16bf afr;
            if (k == 0) {
                const float* b = tx0 + off;
#pragma unroll
                for (int j = 0; j < 8; ++j) {
                    afr[j]     = (__bf16)b[j];
                    afr[j + 8] = (__bf16)b[16 + j];
                }
            } else if (k == 1) {
                const float* b = tx1 + off;
#pragma unroll
                for (int j = 0; j < 8; ++j) {
                    afr[j]     = (__bf16)b[j];
                    afr[j + 8] = (__bf16)b[16 + j];
                }
            } else {  // Tx2 = 2*prop(Tx1) - Tx0, fused on the fly
                const float* bp = p2 + off;
                const float* b0 = tx0 + off;
#pragma unroll
                for (int j = 0; j < 8; ++j) {
                    afr[j]     = (__bf16)(2.0f * bp[j]      - b0[j]);
                    afr[j + 8] = (__bf16)(2.0f * bp[16 + j] - b0[16 + j]);
                }
            }
#pragma unroll
            for (int nt = 0; nt < 4; ++nt) {
                v16bf bfr = *(const v16bf*)(wf + (((k * 2 + kt) * 4 + nt) << 9) + lane * 16);
                acc[nt] = __builtin_amdgcn_wmma_f32_16x16x32_bf16(
                    false, afr, false, bfr, (short)0, acc[nt], false, false);
            }
        }
    }

    // D layout: VGPR r -> m = r + 8*lh, n = lane%16  (ISA 7.12.2 C/D layout)
#pragma unroll
    for (int nt = 0; nt < 4; ++nt) {
#pragma unroll
        for (int r = 0; r < 8; ++r) {
            int m = r + 8 * lh;
            p2[(rt * 16 + m) * 64 + nt * 16 + nlo] = acc[nt][r];
        }
    }
}

// ---------------- final dense head: out[b] = dot(h[b-th graph slab], Wd) + bd --------------
__global__ void dense_kernel(const float* __restrict__ h, const float* __restrict__ Wd,
                             const float* __restrict__ bd, float* __restrict__ out) {
    __shared__ float red[256];
    int b = blockIdx.x;
    const float* base = h + (size_t)b * NUM_NODES * UDIM;  // reshape == contiguous slab
    float s = 0.0f;
    for (int i = threadIdx.x; i < NUM_NODES * UDIM; i += 256)
        s += base[i] * Wd[i];
    red[threadIdx.x] = s;
    __syncthreads();
    for (int off = 128; off > 0; off >>= 1) {
        if (threadIdx.x < off) red[threadIdx.x] += red[threadIdx.x + off];
        __syncthreads();
    }
    if (threadIdx.x == 0) out[b] = red[0] + bd[0];
}

// ---------------- host orchestration ----------------
static inline unsigned gblk(long long n) { return (unsigned)((n + 255) / 256); }

extern "C" void kernel_launch(void* const* d_in, const int* in_sizes, int n_in,
                              void* d_out, int out_size, void* d_ws, size_t ws_size,
                              hipStream_t stream) {
    // dict order: x, edge_index, weights, batch, W1,b1,g1,be1, W2,b2,g2,be2,
    //             W3,b3,g3,be3, W4,b4,g4,be4, Wd,bd
    const float* x   = (const float*)d_in[0];
    const int*   ei  = (const int*)  d_in[1];   // int32 (JAX default x64-off)
    const float* w   = (const float*)d_in[2];
    const float* W1  = (const float*)d_in[4];
    const float* b1  = (const float*)d_in[5];
    const float* g1  = (const float*)d_in[6];
    const float* be1 = (const float*)d_in[7];
    const float* Wl[3]  = {(const float*)d_in[8],  (const float*)d_in[12], (const float*)d_in[16]};
    const float* bl[3]  = {(const float*)d_in[9],  (const float*)d_in[13], (const float*)d_in[17]};
    const float* gl[3]  = {(const float*)d_in[10], (const float*)d_in[14], (const float*)d_in[18]};
    const float* bel[3] = {(const float*)d_in[11], (const float*)d_in[15], (const float*)d_in[19]};
    const float* Wd = (const float*)d_in[20];
    const float* bd = (const float*)d_in[21];
    float* out = (float*)d_out;

    const int* src = ei;
    const int* dst = ei + NE;

    // workspace carve-up (~150 MB); every offset is 256B aligned
    float* ws    = (float*)d_ws;
    float* h     = ws;                    // NN*64
    float* t1    = h   + (size_t)NN * 64; // NN*64
    float* p2    = t1  + (size_t)NN * 64; // NN*64 (prop2 out, then GEMM out in place)
    float* nw    = p2  + (size_t)NN * 64; // NE
    float* deg   = nw  + NE;              // NN (becomes dinv)
    float* t1s   = deg + NN;              // NN*2 (layer-1 prop)
    float* p2s   = t1s + (size_t)NN * 2;  // NN*2
    float* stats = p2s + (size_t)NN * 2;  // 256 floats
    __bf16* wfrag = (__bf16*)(stats + 256); // 9*4096 bf16 fragments

    // ---- prep: weight fragments + edge normalization ----
    wfrag_kernel<<<gblk(9 * 4096), 256, 0, stream>>>(Wl[0], Wl[1], Wl[2], wfrag);
    zero_f32_kernel<<<gblk(NN), 256, 0, stream>>>(deg, NN);
    deg_kernel<<<gblk(NE), 256, 0, stream>>>(src, w, deg, NE);
    dinv_kernel<<<gblk(NN), 256, 0, stream>>>(deg, NN);
    normw_kernel<<<gblk(NE), 256, 0, stream>>>(src, dst, w, deg, nw, NE);

    // ---- layer 1 (C_IN = 2) ----
    zero_f32_kernel<<<gblk((long long)NN * 2), 256, 0, stream>>>(t1s, NN * 2);
    prop_kernel<2><<<gblk((long long)NE * 2), 256, 0, stream>>>(x, t1s, src, dst, nw);
    zero_f32_kernel<<<gblk((long long)NN * 2), 256, 0, stream>>>(p2s, NN * 2);
    prop_kernel<2><<<gblk((long long)NE * 2), 256, 0, stream>>>(t1s, p2s, src, dst, nw);
    lin1_kernel<<<gblk((long long)NN * 64), 256, 0, stream>>>(x, t1s, p2s, W1, b1, p2);
    zero_f32_kernel<<<1, 256, 0, stream>>>(stats, 128);
    colstats_kernel<<<1024, 256, 0, stream>>>(p2, stats);
    bn_finalize_kernel<<<1, 64, 0, stream>>>(stats, g1, be1);
    bn_apply_kernel<<<gblk((long long)NN * 64), 256, 0, stream>>>(p2, h, stats);

    // ---- layers 2..4 (U=64, WMMA path) ----
    for (int l = 0; l < 3; ++l) {
        zero_f32_kernel<<<gblk((long long)NN * 64), 256, 0, stream>>>(t1, NN * 64);
        prop_kernel<64><<<gblk((long long)NE * 64), 256, 0, stream>>>(h, t1, src, dst, nw);
        zero_f32_kernel<<<gblk((long long)NN * 64), 256, 0, stream>>>(p2, NN * 64);
        prop_kernel<64><<<gblk((long long)NE * 64), 256, 0, stream>>>(t1, p2, src, dst, nw);
        // 11392 row tiles / 8 waves per block = 1424 blocks (exact; EXEC stays all-1s)
        cheb_gemm_kernel<<<1424, 256, 0, stream>>>(h, t1, p2, wfrag + (size_t)l * 3 * 4096, bl[l]);
        zero_f32_kernel<<<1, 256, 0, stream>>>(stats, 128);
        colstats_kernel<<<1024, 256, 0, stream>>>(p2, stats);
        bn_finalize_kernel<<<1, 64, 0, stream>>>(stats, gl[l], bel[l]);
        bn_apply_kernel<<<gblk((long long)NN * 64), 256, 0, stream>>>(p2, h, stats);
    }

    // ---- dense head ----
    dense_kernel<<<BGRAPH, 256, 0, stream>>>(h, Wd, bd, out);
}
